// MT5DecoderLayerTransformers_6777458393621
// MI455X (gfx1250) — matrix operand order repaired and checked by reading, not compiled, not run on hardware
//
#include <hip/hip_runtime.h>
#include <hip/hip_bf16.h>
#include <math.h>

// ---------------------------------------------------------------------------
// MT5 decoder self-attention layer for MI455X (gfx1250, wave32, WMMA bf16,
// Tensor Data Mover for all LDS tile staging).
// ---------------------------------------------------------------------------
#define BB 4
#define SS 2048
#define DD 1024
#define HH 16
#define HDIM 64
#define MMROWS (BB * SS) /* 8192 */

typedef __attribute__((ext_vector_type(16))) __bf16 v16bf;
typedef __attribute__((ext_vector_type(8)))  float  v8f;
typedef __attribute__((ext_vector_type(4)))  unsigned int v4u_t;
typedef __attribute__((ext_vector_type(8)))  int v8i_t;
typedef __attribute__((ext_vector_type(4)))  int v4i_t;

union FragU { v16bf v; uint4 u[2]; };

#if defined(__HIP_DEVICE_COMPILE__) && __has_builtin(__builtin_amdgcn_tensor_load_to_lds) && __has_builtin(__builtin_amdgcn_s_wait_tensorcnt)
#define USE_TDM 1
#else
#define USE_TDM 0
#endif

#if USE_TDM
// 2-D tile DMA: global (row_stride_elems, 2-byte elements) -> LDS (packed rows).
// D# layout per CDNA5 ISA 8.3/8.4: group0 = {count, lds_addr, global_addr, type=2},
// group1 = {data_size=2B, tensor dims (huge: tiles are interior), tile dims, stride}.
// This toolchain exposes the 6-arg builtin: (g0 v4u, g1 v8i, g2 v4i, g3 v4i, v8i, cpol).
// Trailing groups are zero (tensor <= 2-D -> groups 2/3 disabled).
__device__ inline void tdm_load_2d(unsigned lds_addr, const void* gptr,
                                   unsigned tile_w_elems, unsigned tile_h,
                                   unsigned long long row_stride_elems) {
    unsigned long long ga = (unsigned long long)(size_t)gptr;
    v4u_t g0;
    g0[0] = 1u;                                                  // count=1, user desc
    g0[1] = lds_addr;                                            // LDS byte offset
    g0[2] = (unsigned)ga;                                        // global_addr[31:0]
    g0[3] = (unsigned)((ga >> 32) & 0x01FFFFFFu) | (2u << 30);   // addr[56:32], type=2
    const unsigned td0 = 0x40000000u, td1 = 0x40000000u;         // no OOB clipping
    v8i_t g1;
    g1[0] = (int)(1u << 16);                                     // wg_mask=0, data_size=2B
    g1[1] = (int)((td0 & 0xFFFFu) << 16);                        // tensor_dim0[15:0]
    g1[2] = (int)((td0 >> 16) | ((td1 & 0xFFFFu) << 16));        // td0 hi | td1 lo
    g1[3] = (int)((td1 >> 16) | (tile_w_elems << 16));           // td1 hi | tile_dim0
    g1[4] = (int)(tile_h & 0xFFFFu);                             // tile_dim1, tile_dim2=0
    g1[5] = (int)(row_stride_elems & 0xFFFFFFFFu);               // dim0 stride lo
    g1[6] = (int)((row_stride_elems >> 32) & 0xFFFFu);           // dim0 stride hi
    g1[7] = 0;
    v4i_t z4 = {0, 0, 0, 0};
    v8i_t z8 = {0, 0, 0, 0, 0, 0, 0, 0};
    __builtin_amdgcn_tensor_load_to_lds(g0, g1, z4, z4, z8, 0);
}
__device__ inline unsigned lds_off(const void* p) { return (unsigned)(size_t)p; }
#endif

// A-operand (16-bit, 16x32) fragment from row-major LDS [rows][pitch]:
// lane<16: row=lane,     K = kb+0..15
// lane>=16: row=lane-16, K = kb+16..31
__device__ inline v16bf ld_a_frag(const __bf16* base, int pitch, int mb, int kb, int lane) {
    int half = lane >> 4;
    int row  = mb + (lane & 15);
    int k0   = kb + half * 16;
    const __bf16* p = base + row * pitch + k0;
    FragU f;
    f.u[0] = *(const uint4*)(p);
    f.u[1] = *(const uint4*)(p + 8);
    return f.v;
}

// B-operand (16-bit, 32x16) fragment from N-major LDS [N][pitchK]:
// lane<16: N=lane, K=kb+0..15 (contiguous 32B); lane>=16: N=lane-16, K=kb+16..31
__device__ inline v16bf ld_b_frag(const __bf16* base, int pitch, int nb, int kb, int lane) {
    int half = lane >> 4;
    int n    = nb + (lane & 15);
    int k0   = kb + half * 16;
    const uint4* p = (const uint4*)(base + n * pitch + k0);
    FragU f;
    f.u[0] = p[0];
    f.u[1] = p[1];
    return f.v;
}

__device__ inline v8f wmma_bf16(v16bf a, v16bf b, v8f c) {
    return __builtin_amdgcn_wmma_f32_16x16x32_bf16(false, a, false, b, (short)0, c, false, false);
}

// ---------------------------------------------------------------------------
// 1) RMS norm (MT5LayerNorm): xn = bf16( x * rsqrt(mean(x^2)+eps) * w )
// ---------------------------------------------------------------------------
__global__ void rmsnorm_bf16_kernel(const float* __restrict__ x,
                                    const float* __restrict__ w,
                                    __bf16* __restrict__ xn) {
    __shared__ float red[8];
    int row = blockIdx.x;
    const float* xr = x + (size_t)row * DD;
    float ss = 0.f;
    for (int i = threadIdx.x; i < DD; i += 256) {
        float v = xr[i];
        ss += v * v;
    }
#pragma unroll
    for (int m = 16; m >= 1; m >>= 1) ss += __shfl_xor(ss, m, 32);
    if ((threadIdx.x & 31) == 0) red[threadIdx.x >> 5] = ss;
    __syncthreads();
    float s2 = 0.f;
#pragma unroll
    for (int i = 0; i < 8; i++) s2 += red[i];
    float rs = rsqrtf(s2 * (1.0f / DD) + 1e-6f);
    for (int i = threadIdx.x; i < DD; i += 256)
        xn[(size_t)row * DD + i] = (__bf16)(xr[i] * rs * w[i]);
}

// ---------------------------------------------------------------------------
// 2) Weight transpose + f32->bf16 convert: wt[n][k] = w[k][n]   (1024x1024)
// ---------------------------------------------------------------------------
__global__ void wtr_kernel(const float* __restrict__ w, __bf16* __restrict__ wt) {
    int idx = blockIdx.x * 256 + threadIdx.x; // over DD*DD
    int k = idx >> 10;
    int n = idx & 1023;
    wt[(size_t)n * DD + k] = (__bf16)w[idx];
}

// ---------------------------------------------------------------------------
// 3) Relative-position bias table: btab[h][d] = rel[bucket(d)][h], d=max(q-k,0)
// ---------------------------------------------------------------------------
__global__ void btab_kernel(const float* __restrict__ rel, float* __restrict__ btab) {
    int idx = blockIdx.x * 256 + threadIdx.x; // over HH*SS
    int h = idx >> 11;
    int d = idx & 2047;
    int bkt;
    if (d < 16) {
        bkt = d;
    } else {
        const float sc = 16.0f / 2.0794415416798357f; // 16 / ln(128/16)
        bkt = 16 + (int)(logf((float)d * (1.0f / 16.0f)) * sc);
        if (bkt > 31) bkt = 31;
    }
    btab[idx] = rel[bkt * HH + h];
}

// ---------------------------------------------------------------------------
// 4) Tiled bf16 GEMM: C[M][N] = A[M][K] * Bt[N][K]^T, K=N=1024
//    epi==0: bf16 store (row-major)
//    epi==1: f32 store of (resid + C)
//    epi==2: bf16 store transposed per-head for V: [B][H][HD][S]
//    Block: 256 threads (8 waves), tile 128x128, BK=32, TDM double-buffered.
// ---------------------------------------------------------------------------
__global__ void gemm_bf16_kernel(const __bf16* __restrict__ A,
                                 const __bf16* __restrict__ Bt,
                                 const float* __restrict__ resid,
                                 void* __restrict__ out,
                                 int epi) {
    __shared__ __bf16 As[2][128 * 32];  // 2 x 8 KB
    __shared__ __bf16 Bs[2][128 * 32];  // 2 x 8 KB
    const int K = DD, N = DD;
    int m0 = blockIdx.y * 128;
    int n0 = blockIdx.x * 128;
    int t = threadIdx.x;
    int lane = t & 31, w = t >> 5;
    int wm = (w & 3) * 32;
    int wn = (w >> 2) * 64;

    v8f zero = {};
    v8f c[2][4];
#pragma unroll
    for (int mi = 0; mi < 2; mi++)
#pragma unroll
        for (int ni = 0; ni < 4; ni++) c[mi][ni] = zero;

#if USE_TDM
    // Prologue: DMA first tile pair.
    if (w == 0) {
        tdm_load_2d(lds_off(&As[0][0]), A + (size_t)m0 * K, 32, 128, K);
        tdm_load_2d(lds_off(&Bs[0][0]), Bt + (size_t)n0 * K, 32, 128, K);
    }
    int buf = 0;
    for (int k0 = 0; k0 < K; k0 += 32) {
        if (w == 0) {
            if (k0 + 32 < K) {
                // Prefetch next tile pair into the other buffer, then wait for
                // the current pair only (TDM completes in order).
                tdm_load_2d(lds_off(&As[buf ^ 1][0]),
                            A + (size_t)m0 * K + (k0 + 32), 32, 128, K);
                tdm_load_2d(lds_off(&Bs[buf ^ 1][0]),
                            Bt + (size_t)n0 * K + (k0 + 32), 32, 128, K);
                __builtin_amdgcn_s_wait_tensorcnt(2);
            } else {
                __builtin_amdgcn_s_wait_tensorcnt(0);
            }
        }
        __syncthreads(); // data ready
        const __bf16* Ab = As[buf];
        const __bf16* Bb = Bs[buf];
        v16bf a0 = ld_a_frag(Ab, 32, wm, 0, lane);
        v16bf a1 = ld_a_frag(Ab, 32, wm + 16, 0, lane);
#pragma unroll
        for (int ni = 0; ni < 4; ni++) {
            v16bf b = ld_b_frag(Bb, 32, wn + ni * 16, 0, lane);
            c[0][ni] = wmma_bf16(a0, b, c[0][ni]);
            c[1][ni] = wmma_bf16(a1, b, c[1][ni]);
        }
        __syncthreads(); // consumption done before this buffer is refilled
        buf ^= 1;
    }
#else
    int lrow = t >> 1;
    int lcol = (t & 1) * 16;
    for (int k0 = 0; k0 < K; k0 += 32) {
        __syncthreads();
        {
            const uint4* ga = (const uint4*)(A + (size_t)(m0 + lrow) * K + k0 + lcol);
            uint4* sa = (uint4*)(&As[0][0] + lrow * 32 + lcol);
            sa[0] = ga[0]; sa[1] = ga[1];
            const uint4* gb = (const uint4*)(Bt + (size_t)(n0 + lrow) * K + k0 + lcol);
            uint4* sb = (uint4*)(&Bs[0][0] + lrow * 32 + lcol);
            sb[0] = gb[0]; sb[1] = gb[1];
        }
        __syncthreads();
        v16bf a0 = ld_a_frag(&As[0][0], 32, wm, 0, lane);
        v16bf a1 = ld_a_frag(&As[0][0], 32, wm + 16, 0, lane);
#pragma unroll
        for (int ni = 0; ni < 4; ni++) {
            v16bf b = ld_b_frag(&Bs[0][0], 32, wn + ni * 16, 0, lane);
            c[0][ni] = wmma_bf16(a0, b, c[0][ni]);
            c[1][ni] = wmma_bf16(a1, b, c[1][ni]);
        }
    }
#endif

    int half = lane >> 4, nl = lane & 15;
#pragma unroll
    for (int mi = 0; mi < 2; mi++)
#pragma unroll
        for (int ni = 0; ni < 4; ni++)
#pragma unroll
            for (int r = 0; r < 8; r++) {
                int row = m0 + wm + mi * 16 + r + 8 * half;
                int col = n0 + wn + ni * 16 + nl;
                float vv = c[mi][ni][r];
                if (epi == 1) {
                    size_t idx = (size_t)row * N + col;
                    ((float*)out)[idx] = resid[idx] + vv;
                } else if (epi == 2) {
                    // V output, transposed per head: vt[b][h][hd][s]
                    int bb = row >> 11, s = row & 2047;
                    int hh = col >> 6, hd = col & 63;
                    ((__bf16*)out)[((size_t)((bb * HH + hh) * HDIM + hd)) * SS + s] = (__bf16)vv;
                } else {
                    ((__bf16*)out)[(size_t)row * N + col] = (__bf16)vv;
                }
            }
}

// ---------------------------------------------------------------------------
// 5) Flash attention per (b, h, 128-row Q tile); key tiles of 64.
//    scores = Q K^T + btab[h][max(q-k,0)], online softmax, ctx = P V.
//    Q/K tiles natural layout [s][hd]; V pre-transposed [hd][s] -> all three
//    tiles are contiguous-row 2-D tiles staged by the TDM.
// ---------------------------------------------------------------------------
__global__ void attn_kernel(const __bf16* __restrict__ Q,
                            const __bf16* __restrict__ Kb,
                            const __bf16* __restrict__ Vt,
                            const float* __restrict__ btab,
                            __bf16* __restrict__ ctx) {
    __shared__ __bf16 Qs[128 * 64];  // 16 KB
    __shared__ __bf16 Ks[64 * 64];   //  8 KB
    __shared__ __bf16 Vts[64 * 64];  //  8 KB ([hd][key])
    __shared__ __bf16 Ps[128 * 64];  // 16 KB

    int q0 = blockIdx.x * 128;
    int h  = blockIdx.y;
    int b  = blockIdx.z;
    int t = threadIdx.x, lane = t & 31, w = t >> 5;
    int half = lane >> 4, nl = lane & 15;
    int mrow = 16 * w;

    const __bf16* qbase = Q + ((size_t)((b * SS + q0) * HH + h)) * HDIM;
#if USE_TDM
    if (w == 0) {
        tdm_load_2d(lds_off(Qs), qbase, 64, 128, HH * HDIM);
        __builtin_amdgcn_s_wait_tensorcnt(0);
    }
#else
    {
        int r = t >> 1, cc = (t & 1) * 32;
        const uint4* g = (const uint4*)(qbase + (size_t)r * HH * HDIM + cc);
        uint4* s = (uint4*)(Qs + r * 64 + cc);
        s[0] = g[0]; s[1] = g[1]; s[2] = g[2]; s[3] = g[3];
    }
#endif
    __syncthreads();
    v16bf qa0 = ld_a_frag(Qs, 64, mrow, 0, lane);
    v16bf qa1 = ld_a_frag(Qs, 64, mrow, 32, lane);

    v8f zero = {};
    v8f o[4];
#pragma unroll
    for (int ni = 0; ni < 4; ni++) o[ni] = zero;
    float mrun[8], lrun[8];
#pragma unroll
    for (int r = 0; r < 8; r++) { mrun[r] = -INFINITY; lrun[r] = 0.f; }

    for (int kt = 0; kt < SS; kt += 64) {
        __syncthreads(); // previous tile fully consumed
#if USE_TDM
        if (w == 0) {
            tdm_load_2d(lds_off(Ks),
                        Kb + ((size_t)((b * SS + kt) * HH + h)) * HDIM,
                        64, 64, HH * HDIM);
            tdm_load_2d(lds_off(Vts),
                        Vt + ((size_t)(b * HH + h)) * HDIM * SS + kt,
                        64, 64, SS);
            __builtin_amdgcn_s_wait_tensorcnt(0);
        }
#else
        {
            int r = t >> 2, cc = (t & 3) * 16;
            const uint4* gk = (const uint4*)(Kb + ((size_t)((b * SS + kt + r) * HH + h)) * HDIM + cc);
            uint4* sk = (uint4*)(Ks + r * 64 + cc);
            sk[0] = gk[0]; sk[1] = gk[1];
            const uint4* gv = (const uint4*)(Vt + ((size_t)((b * HH + h) * HDIM + r)) * SS + kt + cc);
            uint4* sv = (uint4*)(Vts + r * 64 + cc);
            sv[0] = gv[0]; sv[1] = gv[1];
        }
#endif
        __syncthreads();

        // Scores: 4 key-frags x 2 K-steps (HD=64).
        v8f s4[4];
#pragma unroll
        for (int f = 0; f < 4; f++) {
            v16bf bk0 = ld_b_frag(Ks, 64, f * 16, 0, lane);
            v16bf bk1 = ld_b_frag(Ks, 64, f * 16, 32, lane);
            v8f acc = zero;
            acc = wmma_bf16(qa0, bk0, acc);
            acc = wmma_bf16(qa1, bk1, acc);
            s4[f] = acc;
        }

        // Bias + online softmax. Row M = r + 8*half, col N = f*16 + nl.
#pragma unroll
        for (int r = 0; r < 8; r++) {
            int qg = q0 + mrow + r + 8 * half;
            float mx = -INFINITY;
#pragma unroll
            for (int f = 0; f < 4; f++) {
                int kg = kt + f * 16 + nl;
                int d = qg - kg; d = d < 0 ? 0 : d;
                float sv = s4[f][r] + btab[h * SS + d];
                s4[f][r] = sv;
                mx = fmaxf(mx, sv);
            }
#pragma unroll
            for (int m = 8; m >= 1; m >>= 1) mx = fmaxf(mx, __shfl_xor(mx, m, 32));
            float mnew = fmaxf(mrun[r], mx);
            float sc = __expf(mrun[r] - mnew);
            float rsum = 0.f;
#pragma unroll
            for (int f = 0; f < 4; f++) {
                float p = __expf(s4[f][r] - mnew);
                s4[f][r] = p;
                rsum += p;
            }
#pragma unroll
            for (int m = 8; m >= 1; m >>= 1) rsum += __shfl_xor(rsum, m, 32);
            lrun[r] = lrun[r] * sc + rsum;
            mrun[r] = mnew;
#pragma unroll
            for (int ni = 0; ni < 4; ni++) o[ni][r] *= sc;
            // Stage P (C-layout -> LDS row-major) for A-operand reload.
#pragma unroll
            for (int f = 0; f < 4; f++)
                Ps[(mrow + r + 8 * half) * 64 + f * 16 + nl] = (__bf16)s4[f][r];
        }

        // ctx += P (16x64) * V (64x64): 4 hd-frags x 2 K-steps.
        v16bf pa0 = ld_a_frag(Ps, 64, mrow, 0, lane);
        v16bf pa1 = ld_a_frag(Ps, 64, mrow, 32, lane);
#pragma unroll
        for (int ni = 0; ni < 4; ni++) {
            v16bf bv0 = ld_b_frag(Vts, 64, ni * 16, 0, lane);
            v16bf bv1 = ld_b_frag(Vts, 64, ni * 16, 32, lane);
            o[ni] = wmma_bf16(pa0, bv0, o[ni]);
            o[ni] = wmma_bf16(pa1, bv1, o[ni]);
        }
    }

    // Normalize and write ctx as [B][S][H][HD] bf16.
#pragma unroll
    for (int ni = 0; ni < 4; ni++)
#pragma unroll
        for (int r = 0; r < 8; r++) {
            int qg = q0 + mrow + r + 8 * half;
            int hd = ni * 16 + nl;
            float vv = o[ni][r] / lrun[r];
            ctx[((size_t)((b * SS + qg) * HH + h)) * HDIM + hd] = (__bf16)vv;
        }
}

// ---------------------------------------------------------------------------
// Launcher
// ---------------------------------------------------------------------------
extern "C" void kernel_launch(void* const* d_in, const int* in_sizes, int n_in,
                              void* d_out, int out_size, void* d_ws, size_t ws_size,
                              hipStream_t stream) {
    const float* input_ids = (const float*)d_in[0];
    // d_in[1] = encoder_output (unused by reference path)
    const float* wq   = (const float*)d_in[2];
    const float* wk   = (const float*)d_in[3];
    const float* wv   = (const float*)d_in[4];
    const float* wo   = (const float*)d_in[5];
    const float* rel  = (const float*)d_in[6];
    const float* ln_w = (const float*)d_in[7];

    char* ws = (char*)d_ws;
    const size_t SZ_XN  = (size_t)MMROWS * DD * sizeof(__bf16); // 16 MB
    const size_t SZ_W   = (size_t)DD * DD * sizeof(__bf16);     //  2 MB
    const size_t SZ_QKV = (size_t)MMROWS * DD * sizeof(__bf16); // 16 MB
    const size_t SZ_BT  = (size_t)HH * SS * sizeof(float);      // 128 KB

    size_t off = 0;
    __bf16* xn   = (__bf16*)(ws + off); off += SZ_XN;
    __bf16* wqt  = (__bf16*)(ws + off); off += SZ_W;
    __bf16* wkt  = (__bf16*)(ws + off); off += SZ_W;
    __bf16* wvt  = (__bf16*)(ws + off); off += SZ_W;
    __bf16* wot  = (__bf16*)(ws + off); off += SZ_W;
    __bf16* qb   = (__bf16*)(ws + off); off += SZ_QKV;
    __bf16* kb   = (__bf16*)(ws + off); off += SZ_QKV;
    __bf16* vt   = (__bf16*)(ws + off); off += SZ_QKV; // V transposed [B][H][HD][S]
    __bf16* ctxb = (__bf16*)(ws + off); off += SZ_QKV;
    float*  btab = (float*)(ws + off);  off += SZ_BT;
    (void)ws_size; (void)in_sizes; (void)n_in; (void)out_size;

    // 1) RMS norm -> bf16
    rmsnorm_bf16_kernel<<<MMROWS, 256, 0, stream>>>(input_ids, ln_w, xn);

    // 2) Weight transpose/convert
    const int wblocks = (DD * DD) / 256;
    wtr_kernel<<<wblocks, 256, 0, stream>>>(wq, wqt);
    wtr_kernel<<<wblocks, 256, 0, stream>>>(wk, wkt);
    wtr_kernel<<<wblocks, 256, 0, stream>>>(wv, wvt);
    wtr_kernel<<<wblocks, 256, 0, stream>>>(wo, wot);

    // 3) Bias table
    btab_kernel<<<(HH * SS) / 256, 256, 0, stream>>>(rel, btab);

    // 4) QKV projections (V written transposed per head)
    dim3 ggrid(DD / 128, MMROWS / 128);
    gemm_bf16_kernel<<<ggrid, 256, 0, stream>>>(xn, wqt, nullptr, (void*)qb, 0);
    gemm_bf16_kernel<<<ggrid, 256, 0, stream>>>(xn, wkt, nullptr, (void*)kb, 0);
    gemm_bf16_kernel<<<ggrid, 256, 0, stream>>>(xn, wvt, nullptr, (void*)vt, 2);

    // 5) Attention
    dim3 agrid(SS / 128, HH, BB);
    attn_kernel<<<agrid, 256, 0, stream>>>(qb, kb, vt, btab, ctxb);

    // 6) Output projection + residual (f32 out)
    gemm_bf16_kernel<<<ggrid, 256, 0, stream>>>(ctxb, wot, input_ids, d_out, 1);
}
